// MultiHeadAttention_75831942578598
// MI455X (gfx1250) — compile-verified, matrix operand order checked by reading, not run on hardware
//
#include <hip/hip_runtime.h>
#include <hip/hip_bf16.h>

typedef unsigned short u16;
typedef unsigned int   u32;

typedef __attribute__((ext_vector_type(16))) __bf16 v16bf;
typedef __attribute__((ext_vector_type(8)))  float  v8f;
typedef int v4i __attribute__((vector_size(16)));

union Frag {
  v16bf v;
  u16   h[16];
  uint4 q[2];
};

union Pack8 {
  uint4 q;
  u16   h[8];
};

__device__ __forceinline__ u16 f2bf(float f) {
  u32 u = __float_as_uint(f);
  u += 0x7FFFu + ((u >> 16) & 1u);   // round-to-nearest-even
  return (u16)(u >> 16);
}

// -------- async global->LDS (CDNA5) with compile-time fallback --------
#if defined(__has_builtin)
#  if __has_builtin(__builtin_amdgcn_global_load_async_to_lds_b128)
#    define HAVE_ASYNC_LDS 1
#  endif
#endif
#ifndef HAVE_ASYNC_LDS
#  define HAVE_ASYNC_LDS 0
#endif

__device__ __forceinline__ void async_cp16(const u16* g, u16* l) {
#if HAVE_ASYNC_LDS
  __builtin_amdgcn_global_load_async_to_lds_b128(
      (__attribute__((address_space(1))) v4i*)g,
      (__attribute__((address_space(3))) v4i*)l, 0, 0);
#else
  *(uint4*)l = *(const uint4*)g;
#endif
}

__device__ __forceinline__ void wait_async() {
#if HAVE_ASYNC_LDS
#  if __has_builtin(__builtin_amdgcn_s_wait_asynccnt)
  __builtin_amdgcn_s_wait_asynccnt(0);
#  else
  asm volatile("s_wait_asynccnt 0" ::: "memory");
#  endif
#endif
}

#define N_ROWS 8192   // B*T
#define T_SEQ  4096
#define CDIM   512
#define DK     64
#define LDT    72     // padded ushort stride: 144B rows (16B aligned), 36-dword bank walk

// ---------------- fp32 -> bf16 staging ----------------
__global__ void cvt_f32_bf16_kernel(const float* __restrict__ in,
                                    u16* __restrict__ out, int n) {
  int i = blockIdx.x * blockDim.x + threadIdx.x;
  if (i < n) out[i] = f2bf(in[i]);
}

// ---------------- GEMM: out[n,m] = sum_k A[n,k]*W[m,k] + bias[m] ----------------
// Block tile 64x128, 4 waves in 2x2, each wave 32x64 (2x4 WMMA accumulators).
// OUTMODE: 0 = bf16 [N,512], 1 = f32 [N,512], 2 = bf16 transposed [512,N] (for V^T)
template <int OUTMODE>
__global__ __launch_bounds__(128)
void gemm512_bias_kernel(const u16* __restrict__ A,
                         const u16* __restrict__ W,
                         const float* __restrict__ bias,
                         u16* __restrict__ outB, float* __restrict__ outF) {
  __shared__ __align__(16) u16 As[2][64 * LDT];    // 64 rows x 32 k
  __shared__ __align__(16) u16 Bs[2][128 * LDT];   // 128 cols x 32 k

  const int rowBase = blockIdx.x * 64;
  const int colBase = blockIdx.y * 128;
  const int t    = threadIdx.x;
  const int wave = t >> 5;
  const int lane = t & 31;
  const int lm   = lane & 15;
  const int half = lane >> 4;
  const int wr   = wave >> 1;   // wave row half (0/1) -> 32 rows
  const int wc   = wave & 1;    // wave col half (0/1) -> 64 cols

  v8f acc[2][4] = {};

  // A stage: 64x32 = 2048 elems, thread: row t>>1, 16 elems at (t&1)*16
  const int aRow = t >> 1, aCg = (t & 1) * 16;
  // B stage: 128x32 = 4096 elems, thread: row t, 32 elems
  const u16* gA = A + (size_t)(rowBase + aRow) * CDIM + aCg;
  const u16* gW = W + (size_t)(colBase + t) * CDIM;
  u16* sAp = &As[0][0] + aRow * LDT + aCg;
  u16* sBp = &Bs[0][0] + t * LDT;
  const int aStride = 64 * LDT, bStride = 128 * LDT;

  // prologue: stage k0 = 0 into buffer 0
#if HAVE_ASYNC_LDS
  {
    for (int i = 0; i < 2; ++i) async_cp16(gA + i * 8, sAp + i * 8);
    for (int i = 0; i < 4; ++i) async_cp16(gW + i * 8, sBp + i * 8);
    wait_async();
  }
#else
  {
    uint4 ra0 = ((const uint4*)gA)[0], ra1 = ((const uint4*)gA)[1];
    uint4 rb0 = ((const uint4*)gW)[0], rb1 = ((const uint4*)gW)[1];
    uint4 rb2 = ((const uint4*)gW)[2], rb3 = ((const uint4*)gW)[3];
    ((uint4*)sAp)[0] = ra0; ((uint4*)sAp)[1] = ra1;
    ((uint4*)sBp)[0] = rb0; ((uint4*)sBp)[1] = rb1;
    ((uint4*)sBp)[2] = rb2; ((uint4*)sBp)[3] = rb3;
  }
#endif

  int buf = 0;
  for (int k0 = 0; k0 < CDIM; k0 += 32) {
    const bool hasNext = (k0 + 32) < CDIM;
#if !HAVE_ASYNC_LDS
    uint4 ra0, ra1, rb0, rb1, rb2, rb3;
    if (hasNext) {   // issue next tile's global loads first (overlap with WMMA)
      const u16* pa = gA + k0 + 32;
      const u16* pw = gW + k0 + 32;
      ra0 = ((const uint4*)pa)[0]; ra1 = ((const uint4*)pa)[1];
      rb0 = ((const uint4*)pw)[0]; rb1 = ((const uint4*)pw)[1];
      rb2 = ((const uint4*)pw)[2]; rb3 = ((const uint4*)pw)[3];
    }
#endif
    __syncthreads();   // current buffer complete & visible; prior readers of buf^1 done
#if HAVE_ASYNC_LDS
    if (hasNext) {     // fused async copies into the other buffer (ASYNCcnt)
      const u16* pa = gA + k0 + 32;
      const u16* pw = gW + k0 + 32;
      u16* da = sAp + (buf ^ 1) * aStride;
      u16* db = sBp + (buf ^ 1) * bStride;
      for (int i = 0; i < 2; ++i) async_cp16(pa + i * 8, da + i * 8);
      for (int i = 0; i < 4; ++i) async_cp16(pw + i * 8, db + i * 8);
    }
#endif

    Frag a[2], b[4];
    const u16* ab = &As[buf][0];
    const u16* bb = &Bs[buf][0];
    for (int mi = 0; mi < 2; ++mi) {
      const u16* p = ab + (wr * 32 + mi * 16 + lm) * LDT;
      a[mi].q[0] = *(const uint4*)(p + half * 8);
      a[mi].q[1] = *(const uint4*)(p + 16 + half * 8);
    }
    for (int ni = 0; ni < 4; ++ni) {
      const u16* p = bb + (wc * 64 + ni * 16 + lm) * LDT;
      b[ni].q[0] = *(const uint4*)(p + half * 8);
      b[ni].q[1] = *(const uint4*)(p + 16 + half * 8);
    }
    for (int mi = 0; mi < 2; ++mi)
      for (int ni = 0; ni < 4; ++ni)
        acc[mi][ni] = __builtin_amdgcn_wmma_f32_16x16x32_bf16(
            false, a[mi].v, false, b[ni].v, (short)0, acc[mi][ni], false, false);

#if HAVE_ASYNC_LDS
    if (hasNext) wait_async();   // own async copies done before next barrier
#else
    if (hasNext) {   // park staged regs into the other buffer
      u16* da = sAp + (buf ^ 1) * aStride;
      u16* db = sBp + (buf ^ 1) * bStride;
      ((uint4*)da)[0] = ra0; ((uint4*)da)[1] = ra1;
      ((uint4*)db)[0] = rb0; ((uint4*)db)[1] = rb1;
      ((uint4*)db)[2] = rb2; ((uint4*)db)[3] = rb3;
    }
#endif
    buf ^= 1;
  }

  for (int mi = 0; mi < 2; ++mi)
    for (int ni = 0; ni < 4; ++ni) {
      const int col = colBase + wc * 64 + ni * 16 + lm;
      const float bv = bias[col];
      const int row0 = rowBase + wr * 32 + mi * 16 + half * 8;
      if (OUTMODE == 2) {
        Pack8 pk;
        for (int r = 0; r < 8; ++r) pk.h[r] = f2bf(acc[mi][ni][r] + bv);
        *(uint4*)(outB + (size_t)col * N_ROWS + row0) = pk.q;   // V^T: rows contiguous
      } else {
        for (int r = 0; r < 8; ++r) {
          const float val = acc[mi][ni][r] + bv;
          if (OUTMODE == 1) outF[(size_t)(row0 + r) * CDIM + col] = val;
          else              outB[(size_t)(row0 + r) * CDIM + col] = f2bf(val);
        }
      }
    }
}

// ---------------- Flash attention: per (b, h, 64 q-rows), 64-key tiles ----------------
// K:  [N_ROWS, 512] bf16 (d contiguous).  Vt: [512, N_ROWS] bf16 (tokens contiguous).
__global__ __launch_bounds__(128)
void flash_attn_kernel(const u16* __restrict__ Q, const u16* __restrict__ K,
                       const u16* __restrict__ Vt, u16* __restrict__ O) {
  __shared__ __align__(16) u16 Ks[2][64 * LDT];   // [j][d]  64 keys x 64 d
  __shared__ __align__(16) u16 Vs[2][64 * LDT];   // [d][j]  64 d x 64 keys (transposed V)
  __shared__ __align__(16) u16 Ps[4 * 16 * LDT];  // per-wave P (16 x 64) relayout scratch

  const int qt   = blockIdx.x;
  const int h    = blockIdx.y;
  const int b    = blockIdx.z;
  const int t    = threadIdx.x;
  const int wave = t >> 5;
  const int lane = t & 31;
  const int lm   = lane & 15;
  const int half = lane >> 4;
  const int hc   = h * DK;
  const size_t nb = (size_t)b * T_SEQ;

  // Q fragments: this wave's 16 q-rows, d=0..63 split into two k=32 fragments
  Frag qf[2];
  {
    const u16* qp = Q + (nb + qt * 64 + wave * 16 + lm) * CDIM + hc;
    for (int ki = 0; ki < 2; ++ki) {
      qf[ki].q[0] = *(const uint4*)(qp + ki * 32 + half * 8);
      qf[ki].q[1] = *(const uint4*)(qp + ki * 32 + 16 + half * 8);
    }
  }

  v8f oacc[4] = {};
  float mrow[8], lrow[8];
  for (int r = 0; r < 8; ++r) { mrow[r] = -1e30f; lrow[r] = 0.f; }

  // staging: both tiles are 64x64 = 4096 elems; thread: row t>>1, 32 elems at (t&1)*32
  const int sRow = t >> 1, sCg = (t & 1) * 32;
  const u16* gK = K + (nb + sRow) * CDIM + hc + sCg;
  const u16* gV = Vt + (size_t)(hc + sRow) * N_ROWS + nb + sCg;
  u16* sK = &Ks[0][0] + sRow * LDT + sCg;
  u16* sV = &Vs[0][0] + sRow * LDT + sCg;
  const int tStride = 64 * LDT;

  // prologue: stage j0 = 0 into buffer 0
#if HAVE_ASYNC_LDS
  {
    for (int i = 0; i < 4; ++i) async_cp16(gK + i * 8, sK + i * 8);
    for (int i = 0; i < 4; ++i) async_cp16(gV + i * 8, sV + i * 8);
    wait_async();
  }
#else
  {
    uint4 rk[4], rv[4];
    for (int i = 0; i < 4; ++i) { rk[i] = ((const uint4*)gK)[i]; rv[i] = ((const uint4*)gV)[i]; }
    for (int i = 0; i < 4; ++i) { ((uint4*)sK)[i] = rk[i]; ((uint4*)sV)[i] = rv[i]; }
  }
#endif

  int buf = 0;
  for (int j0 = 0; j0 < T_SEQ; j0 += 64) {
    const bool hasNext = (j0 + 64) < T_SEQ;
#if !HAVE_ASYNC_LDS
    uint4 rk[4], rv[4];
    if (hasNext) {   // issue next K/V tile loads before compute
      const u16* pk = gK + (size_t)(j0 + 64) * CDIM;
      const u16* pv = gV + (j0 + 64);
      for (int i = 0; i < 4; ++i) { rk[i] = ((const uint4*)pk)[i]; rv[i] = ((const uint4*)pv)[i]; }
    }
#endif
    if (j0 + 128 < T_SEQ) {   // warm L2/WGP$ for the tile after next
      __builtin_prefetch(gK + (size_t)(j0 + 128) * CDIM, 0, 1);
      __builtin_prefetch(gV + (j0 + 128), 0, 1);
    }
    __syncthreads();
#if HAVE_ASYNC_LDS
    if (hasNext) {   // fused async copies into the other buffer (ASYNCcnt)
      const u16* pk = gK + (size_t)(j0 + 64) * CDIM;
      const u16* pv = gV + (j0 + 64);
      u16* dk = sK + (buf ^ 1) * tStride;
      u16* dv = sV + (buf ^ 1) * tStride;
      for (int i = 0; i < 4; ++i) async_cp16(pk + i * 8, dk + i * 8);
      for (int i = 0; i < 4; ++i) async_cp16(pv + i * 8, dv + i * 8);
    }
#endif

    // S[16 x 64] = Q_tile (16x64) . K_tile^T : four 16-col subtiles, two k-steps each
    const u16* kb = &Ks[buf][0];
    const u16* vb = &Vs[buf][0];
    v8f s[4] = {};
    for (int js = 0; js < 4; ++js)
      for (int ki = 0; ki < 2; ++ki) {
        Frag kf;
        const u16* kp = kb + (js * 16 + lm) * LDT + ki * 32;
        kf.q[0] = *(const uint4*)(kp + half * 8);
        kf.q[1] = *(const uint4*)(kp + 16 + half * 8);
        s[js] = __builtin_amdgcn_wmma_f32_16x16x32_bf16(
            false, qf[ki].v, false, kf.v, (short)0, s[js], false, false);
      }

    // Online softmax over the 64-key tile (row stats via 16-lane shfl halves)
    float pm[8], ps[8];
    for (int r = 0; r < 8; ++r) {
      s[0][r] *= 0.125f; s[1][r] *= 0.125f;   // 1/sqrt(64)
      s[2][r] *= 0.125f; s[3][r] *= 0.125f;
      pm[r] = fmaxf(fmaxf(s[0][r], s[1][r]), fmaxf(s[2][r], s[3][r]));
    }
    for (int m = 1; m < 16; m <<= 1)
      for (int r = 0; r < 8; ++r)
        pm[r] = fmaxf(pm[r], __shfl_xor(pm[r], m));
    for (int r = 0; r < 8; ++r) {
      const float mn = fmaxf(mrow[r], pm[r]);
      const float al = __expf(mrow[r] - mn);
      mrow[r] = mn;
      s[0][r] = __expf(s[0][r] - mn);
      s[1][r] = __expf(s[1][r] - mn);
      s[2][r] = __expf(s[2][r] - mn);
      s[3][r] = __expf(s[3][r] - mn);
      ps[r] = (s[0][r] + s[1][r]) + (s[2][r] + s[3][r]);
      lrow[r] *= al;
      for (int d = 0; d < 4; ++d) oacc[d][r] *= al;
    }
    for (int m = 1; m < 16; m <<= 1)
      for (int r = 0; r < 8; ++r)
        ps[r] += __shfl_xor(ps[r], m);
    for (int r = 0; r < 8; ++r) lrow[r] += ps[r];

    // P (16x64): C-layout fp32 -> bf16 A-layout via per-wave LDS (same-wave DS in-order)
    u16* pw = Ps + wave * 16 * LDT;
    for (int r = 0; r < 8; ++r) {
      const int mi = half * 8 + r;
      for (int js = 0; js < 4; ++js)
        pw[mi * LDT + js * 16 + lm] = f2bf(s[js][r]);
    }
    Frag pf[2];
    {
      const u16* pp = pw + lm * LDT;
      for (int ki = 0; ki < 2; ++ki) {
        pf[ki].q[0] = *(const uint4*)(pp + ki * 32 + half * 8);
        pf[ki].q[1] = *(const uint4*)(pp + ki * 32 + 16 + half * 8);
      }
    }

    // O += P (16x64) . V_tile (64x64): V^T tile is [d][j], two k-steps per d-subtile
    for (int d = 0; d < 4; ++d) {
      const u16* vp = vb + (d * 16 + lm) * LDT;
      for (int ki = 0; ki < 2; ++ki) {
        Frag vf;
        vf.q[0] = *(const uint4*)(vp + ki * 32 + half * 8);
        vf.q[1] = *(const uint4*)(vp + ki * 32 + 16 + half * 8);
        oacc[d] = __builtin_amdgcn_wmma_f32_16x16x32_bf16(
            false, pf[ki].v, false, vf.v, (short)0, oacc[d], false, false);
      }
    }

#if HAVE_ASYNC_LDS
    if (hasNext) wait_async();   // own async copies done before next barrier
#else
    if (hasNext) {
      u16* dk = sK + (buf ^ 1) * tStride;
      u16* dv = sV + (buf ^ 1) * tStride;
      for (int i = 0; i < 4; ++i) { ((uint4*)dk)[i] = rk[i]; ((uint4*)dv)[i] = rv[i]; }
    }
#endif
    buf ^= 1;
  }

  for (int d = 0; d < 4; ++d)
    for (int r = 0; r < 8; ++r) {
      const size_t row = nb + qt * 64 + wave * 16 + half * 8 + r;
      O[row * CDIM + hc + d * 16 + lm] = f2bf(oacc[d][r] / lrow[r]);
    }
}

extern "C" void kernel_launch(void* const* d_in, const int* in_sizes, int n_in,
                              void* d_out, int out_size, void* d_ws, size_t ws_size,
                              hipStream_t stream) {
  const float* x  = (const float*)d_in[0];
  const float* Wq = (const float*)d_in[1];
  const float* bq = (const float*)d_in[2];
  const float* Wk = (const float*)d_in[3];
  const float* bk = (const float*)d_in[4];
  const float* Wv = (const float*)d_in[5];
  const float* bv = (const float*)d_in[6];
  const float* Wo = (const float*)d_in[7];
  const float* bo = (const float*)d_in[8];
  float* out = (float*)d_out;

  char* ws = (char*)d_ws;
  const size_t szAct = (size_t)N_ROWS * CDIM * sizeof(u16);  // 8 MB
  const size_t szW   = (size_t)CDIM * CDIM * sizeof(u16);    // 512 KB
  u16* xb  = (u16*)ws; ws += szAct;
  u16* qb  = (u16*)ws; ws += szAct;
  u16* kb  = (u16*)ws; ws += szAct;
  u16* vtb = (u16*)ws; ws += szAct;   // V^T: [512, N_ROWS]
  u16* ob  = (u16*)ws; ws += szAct;
  u16* wqb = (u16*)ws; ws += szW;
  u16* wkb = (u16*)ws; ws += szW;
  u16* wvb = (u16*)ws; ws += szW;
  u16* wob = (u16*)ws; ws += szW;

  const int nAct = N_ROWS * CDIM;
  const int nW   = CDIM * CDIM;
  cvt_f32_bf16_kernel<<<(nAct + 255) / 256, 256, 0, stream>>>(x,  xb,  nAct);
  cvt_f32_bf16_kernel<<<(nW + 255) / 256,  256, 0, stream>>>(Wq, wqb, nW);
  cvt_f32_bf16_kernel<<<(nW + 255) / 256,  256, 0, stream>>>(Wk, wkb, nW);
  cvt_f32_bf16_kernel<<<(nW + 255) / 256,  256, 0, stream>>>(Wv, wvb, nW);
  cvt_f32_bf16_kernel<<<(nW + 255) / 256,  256, 0, stream>>>(Wo, wob, nW);

  dim3 gGemm(N_ROWS / 64, CDIM / 128);
  gemm512_bias_kernel<0><<<gGemm, 128, 0, stream>>>(xb, wqb, bq, qb, nullptr);
  gemm512_bias_kernel<0><<<gGemm, 128, 0, stream>>>(xb, wkb, bk, kb, nullptr);
  gemm512_bias_kernel<2><<<gGemm, 128, 0, stream>>>(xb, wvb, bv, vtb, nullptr);

  dim3 gFlash(T_SEQ / 64, 8, 2);
  flash_attn_kernel<<<gFlash, 128, 0, stream>>>(qb, kb, vtb, ob);

  gemm512_bias_kernel<1><<<gGemm, 128, 0, stream>>>(ob, wob, bo, nullptr, out);
}
